// AFNONet_24507083391113
// MI455X (gfx1250) — matrix operand order repaired and checked by reading, not compiled, not run on hardware
//
#include <hip/hip_runtime.h>

typedef __attribute__((ext_vector_type(16))) _Float16 v16h;
typedef __attribute__((ext_vector_type(8)))  _Float16 h8;
typedef __attribute__((ext_vector_type(8)))  float    v8f;

namespace {

constexpr int cHG = 45, cWG = 90, cTOK = cHG * cWG;   // 4050 tokens
constexpr int cE  = 768, cMLP = 3072;
constexpr int cKW = 23;                               // kept W-frequencies (23 of 46 rfft bins)
constexpr int cPTS  = cHG * cKW;                      // 1035 frequency points
constexpr int cPTSP = 1040;                           // padded to multiple of 16
constexpr int cNB = 16, cBS = 48, cCATK = 96;         // freq block MLP: K = [re|im] = 96
constexpr int cHI = 720, cWI = 1440;
constexpr long long cHWI = (long long)cHI * cWI;
constexpr int cPK = 1280;                             // 5 ch * 16 * 16 patch K
constexpr float cLAM = 0.01f;                         // softshrink lambda (SPARSITY)

// --------------------------------------------------------------------------
// Generic f16 WMMA GEMM: C[M,N] = act(A[M,K] @ B[K,N] + bias) + add
// Workgroup tile 128x64, 8 waves, wave tile 32x32 (4 x v_wmma_f32_16x16x32_f16
// per K-step from 2 A-frags x 2 B-frags). Double-buffered LDS, one barrier
// per K-step; next tile's global loads issue before current tile's WMMAs.
// Optional f32 (Cf) and/or f16 (Ch) outputs. Batched over blockIdx.z.
// act: 0 none, 1 relu, 2 exact gelu, 3 softshrink(0.01)
// Requires: K % 32 == 0, N % 8 == 0, 16B-aligned rows (true for all call sites).
// --------------------------------------------------------------------------
__global__ __launch_bounds__(256)
void k_wmma_gemm(const _Float16* __restrict__ A, const _Float16* __restrict__ B,
                 float* Cf, _Float16* Ch,
                 const float* __restrict__ bias, const float* add,
                 int M, int N, int K, int lda, int ldb, int ldc, int ldadd,
                 long long sA, long long sB, long long sC, long long sBias,
                 int act)
{
  const long long bz = blockIdx.z;
  A += bz * sA;  B += bz * sB;
  if (Cf)   Cf   += bz * sC;
  if (Ch)   Ch   += bz * sC;
  if (bias) bias += bz * sBias;

  const int tid  = threadIdx.x;
  const int lane = tid & 31, wid = tid >> 5;
  const int bm = blockIdx.y * 128, bn = blockIdx.x * 64;

  __shared__ alignas(16) _Float16 As[2][128][32];   // [buf][m][k]
  __shared__ alignas(16) _Float16 Bs[2][64][32];    // [buf][n][k] (transposed)

  v8f a00, a01, a10, a11;
  #pragma unroll
  for (int i = 0; i < 8; ++i) { a00[i] = 0.f; a01[i] = 0.f; a10[i] = 0.f; a11[i] = 0.f; }

  // wave -> 32x32 tile: 4 m-quarters x 2 n-halves
  const int m0 = (wid & 3) * 32;
  const int n0 = (wid >> 2) * 32;
  const int lr = lane & 15;
  const int kh = (lane >> 4) * 8;               // K-half select per ISA A layout

  // A loader: 256 thr x 16 halves = 128x32 tile. row 0..127, col-seg {0,16}
  const int ar = tid >> 1,  ac = (tid & 1) * 16;
  // B loader: 256 thr x 8 halves = 32x64 tile. k-row 0..31, col-seg
  const int br = tid >> 3,  bc = (tid & 7) * 8;

  const int  gmA = bm + ar;
  const int  gnB = bn + bc;
  const bool aok = gmA < M;
  const bool bok = gnB < N;

  h8 av0, av1, bv;
  #pragma unroll
  for (int j = 0; j < 8; ++j) { av0[j] = (_Float16)0.f; av1[j] = (_Float16)0.f; bv[j] = (_Float16)0.f; }

  // ---- prologue: fetch + stage tile 0 ----
  if (aok) {
    av0 = *(const h8*)(A + (long long)gmA * lda + ac);
    av1 = *(const h8*)(A + (long long)gmA * lda + ac + 8);
  }
  if (bok) bv = *(const h8*)(B + (long long)br * ldb + gnB);
  *(h8*)(&As[0][ar][ac])     = av0;
  *(h8*)(&As[0][ar][ac + 8]) = av1;
  #pragma unroll
  for (int j = 0; j < 8; ++j) Bs[0][bc + j][br] = bv[j];
  __syncthreads();

  const int nk = K >> 5;
  for (int kt = 0; kt < nk; ++kt) {
    const int cur = kt & 1, nxt = cur ^ 1;
    const bool has_next = (kt + 1) < nk;

    // ---- issue next tile's global loads first (latency hides under WMMA) ----
    if (has_next) {
      const int k1 = (kt + 1) << 5;
      if (aok) {
        av0 = *(const h8*)(A + (long long)gmA * lda + k1 + ac);
        av1 = *(const h8*)(A + (long long)gmA * lda + k1 + ac + 8);
      }
      if (bok) bv = *(const h8*)(B + (long long)(k1 + br) * ldb + gnB);
    }

    // ---- compute from current buffer ----
    // Fragments per ISA 7.12.2 16-bit layout:
    // lanes 0-15: K in {0..7} U {16..23}; lanes 16-31: K in {8..15} U {24..31}
    v16h af0, af1, bf0, bf1;
    {
      h8 lo = *(const h8*)(&As[cur][m0 + lr][kh]);
      h8 hi = *(const h8*)(&As[cur][m0 + lr][16 + kh]);
      #pragma unroll
      for (int j = 0; j < 8; ++j) { af0[j] = lo[j]; af0[8 + j] = hi[j]; }
      lo = *(const h8*)(&As[cur][m0 + 16 + lr][kh]);
      hi = *(const h8*)(&As[cur][m0 + 16 + lr][16 + kh]);
      #pragma unroll
      for (int j = 0; j < 8; ++j) { af1[j] = lo[j]; af1[8 + j] = hi[j]; }
    }
    {
      h8 lo = *(const h8*)(&Bs[cur][n0 + lr][kh]);
      h8 hi = *(const h8*)(&Bs[cur][n0 + lr][16 + kh]);
      #pragma unroll
      for (int j = 0; j < 8; ++j) { bf0[j] = lo[j]; bf0[8 + j] = hi[j]; }
      lo = *(const h8*)(&Bs[cur][n0 + 16 + lr][kh]);
      hi = *(const h8*)(&Bs[cur][n0 + 16 + lr][16 + kh]);
      #pragma unroll
      for (int j = 0; j < 8; ++j) { bf1[j] = lo[j]; bf1[8 + j] = hi[j]; }
    }
    a00 = __builtin_amdgcn_wmma_f32_16x16x32_f16(false, af0, false, bf0, (short)0, a00, false, false);
    a01 = __builtin_amdgcn_wmma_f32_16x16x32_f16(false, af0, false, bf1, (short)0, a01, false, false);
    a10 = __builtin_amdgcn_wmma_f32_16x16x32_f16(false, af1, false, bf0, (short)0, a10, false, false);
    a11 = __builtin_amdgcn_wmma_f32_16x16x32_f16(false, af1, false, bf1, (short)0, a11, false, false);

    // ---- stage next tile into the other buffer ----
    if (has_next) {
      *(h8*)(&As[nxt][ar][ac])     = av0;
      *(h8*)(&As[nxt][ar][ac + 8]) = av1;
      #pragma unroll
      for (int j = 0; j < 8; ++j) Bs[nxt][bc + j][br] = bv[j];
    }
    __syncthreads();
  }

  // D layout: VGPR i, lanes 0-15 -> M=i, N=lane; lanes 16-31 -> M=i+8, N=lane-16
  const int rofs = (lane >> 4) * 8;
  #pragma unroll
  for (int mi = 0; mi < 2; ++mi) {
    #pragma unroll
    for (int i = 0; i < 8; ++i) {
      const int gm = bm + m0 + mi * 16 + i + rofs;
      if (gm >= M) continue;
      #pragma unroll
      for (int ni = 0; ni < 2; ++ni) {
        const int gn = bn + n0 + ni * 16 + lr;
        if (gn >= N) continue;
        float v = (mi == 0) ? ((ni == 0) ? a00[i] : a01[i])
                            : ((ni == 0) ? a10[i] : a11[i]);
        if (bias) v += bias[gn];
        if (act == 1)      v = fmaxf(v, 0.f);
        else if (act == 2) v = 0.5f * v * (1.f + erff(v * 0.70710678118654752f));
        else if (act == 3) v = (v > cLAM) ? (v - cLAM) : ((v < -cLAM) ? (v + cLAM) : 0.f);
        if (add) v += add[(long long)gm * ldadd + gn];
        if (Cf) Cf[(long long)gm * ldc + gn] = v;
        if (Ch) Ch[(long long)gm * ldc + gn] = (_Float16)v;
      }
    }
  }
}

// -------------------------- LayerNorm over last dim (768) -----------------
__global__ __launch_bounds__(256)
void k_layernorm(const float* __restrict__ x, const float* __restrict__ g,
                 const float* __restrict__ b, float* __restrict__ y, _Float16* y16)
{
  const int t = blockIdx.x;
  const float* xr = x + (long long)t * cE;
  float s = 0.f, s2 = 0.f;
  for (int i = threadIdx.x; i < cE; i += 256) { float v = xr[i]; s += v; s2 += v * v; }
  __shared__ float sh[256], sh2[256];
  sh[threadIdx.x] = s; sh2[threadIdx.x] = s2;
  __syncthreads();
  for (int st = 128; st > 0; st >>= 1) {
    if (threadIdx.x < st) { sh[threadIdx.x] += sh[threadIdx.x + st];
                            sh2[threadIdx.x] += sh2[threadIdx.x + st]; }
    __syncthreads();
  }
  const float mean = sh[0] * (1.f / cE);
  const float var  = sh2[0] * (1.f / cE) - mean * mean;
  const float r = rsqrtf(var + 1e-6f);
  for (int i = threadIdx.x; i < cE; i += 256) {
    float v = (xr[i] - mean) * r * g[i] + b[i];
    y[(long long)t * cE + i] = v;
    if (y16) y16[(long long)t * cE + i] = (_Float16)v;
  }
}

// -------------------------- DFT twiddle tables (ortho norm) ---------------
__global__ __launch_bounds__(256)
void k_init_tables(float* fwc, float* fws, float* fhc, float* fhs, float* iwc, float* iws)
{
  int i = blockIdx.x * 256 + threadIdx.x;
  const float is90 = rsqrtf(90.f), is45 = rsqrtf(45.f);
  if (i < cKW * cWG) {
    int kw = i / cWG, w = i % cWG;
    float th = 6.283185307179586f * (float)(kw * w) / 90.f;
    float c = cosf(th), sn = sinf(th);
    fwc[i] = c * is90;  fws[i] = -sn * is90;                   // e^{-i th}
    iwc[i] = ((kw == 0) ? 1.f : 2.f) * c * is90;               // irfft (Hermitian fold)
    iws[i] = (kw == 0) ? 0.f : (-2.f * sn * is90);
  }
  if (i < cHG * cHG) {
    int kh = i / cHG, h = i % cHG;
    float th = 6.283185307179586f * (float)(kh * h) / 45.f;
    fhc[i] = cosf(th) * is45;  fhs[i] = sinf(th) * is45;
  }
}

// Forward rDFT along W: x[h][w][c] -> Y[h][kw][c] (complex)
__global__ __launch_bounds__(256)
void k_fft_w_fwd(const float* __restrict__ xln, const float* __restrict__ fwc,
                 const float* __restrict__ fws, float* __restrict__ yr, float* __restrict__ yi)
{
  int idx = blockIdx.x * 256 + threadIdx.x;
  if (idx >= cPTS * cE) return;
  int c = idx % cE, kw = (idx / cE) % cKW, h = idx / (cE * cKW);
  const float* xrow = xln + (long long)h * cWG * cE + c;
  float ar = 0.f, ai = 0.f;
  for (int w = 0; w < cWG; ++w) {
    float v = xrow[(long long)w * cE];
    ar = fmaf(v, fwc[kw * cWG + w], ar);
    ai = fmaf(v, fws[kw * cWG + w], ai);
  }
  yr[idx] = ar; yi[idx] = ai;
}

// Forward DFT along H; write f16 [re|im] concat layout [pt][n][96] for WMMA
__global__ __launch_bounds__(256)
void k_fft_h_fwd(const float* __restrict__ yr, const float* __restrict__ yi,
                 const float* __restrict__ fhc, const float* __restrict__ fhs,
                 _Float16* __restrict__ xcat)
{
  int idx = blockIdx.x * 256 + threadIdx.x;
  if (idx >= cPTS * cE) return;
  int c = idx % cE, kw = (idx / cE) % cKW, kh = idx / (cE * cKW);
  float zr = 0.f, zi = 0.f;
  for (int h = 0; h < cHG; ++h) {
    long long o = ((long long)h * cKW + kw) * cE + c;
    float a = yr[o], b = yi[o];
    float cc = fhc[kh * cHG + h], ss = fhs[kh * cHG + h];
    zr += a * cc + b * ss;    // (a+ib)(cc - i ss)
    zi += b * cc - a * ss;
  }
  int n = c / cBS, j = c % cBS;
  long long base = ((long long)kh * cKW + kw) * (cNB * cCATK) + (long long)n * cCATK + j;
  xcat[base]       = (_Float16)zr;
  xcat[base + cBS] = (_Float16)zi;
}

// Inverse DFT along H
__global__ __launch_bounds__(256)
void k_fft_h_inv(const float* __restrict__ o2r, const float* __restrict__ o2i,
                 const float* __restrict__ fhc, const float* __restrict__ fhs,
                 float* __restrict__ ur, float* __restrict__ ui)
{
  int idx = blockIdx.x * 256 + threadIdx.x;
  if (idx >= cPTS * cE) return;
  int c = idx % cE, kw = (idx / cE) % cKW, h = idx / (cE * cKW);
  float ar = 0.f, ai = 0.f;
  for (int kh = 0; kh < cHG; ++kh) {
    long long o = ((long long)kh * cKW + kw) * cE + c;
    float zr = o2r[o], zi = o2i[o];
    float cc = fhc[kh * cHG + h], ss = fhs[kh * cHG + h];
    ar += zr * cc - zi * ss;  // (zr+izi)(cc + i ss)
    ai += zi * cc + zr * ss;
  }
  ur[idx] = ar; ui[idx] = ai;
}

// Inverse rDFT along W + AFNO bias (LN1 output) + block residual (old x)
__global__ __launch_bounds__(256)
void k_fft_w_inv(const float* __restrict__ ur, const float* __restrict__ ui,
                 const float* __restrict__ iwc, const float* __restrict__ iws,
                 const float* __restrict__ xln, float* x)
{
  int idx = blockIdx.x * 256 + threadIdx.x;
  if (idx >= cTOK * cE) return;
  int c = idx % cE, w = (idx / cE) % cWG, h = idx / (cE * cWG);
  float acc = 0.f;
  for (int kw = 0; kw < cKW; ++kw) {
    long long o = ((long long)h * cKW + kw) * cE + c;
    acc += ur[o] * iwc[kw * cWG + w] + ui[o] * iws[kw * cWG + w];
  }
  x[idx] = acc + xln[idx] + x[idx];
}

// Build [w; +/-w] concatenated f16 weights for the frequency block-MLP
__global__ __launch_bounds__(256)
void k_build_cats(const float* __restrict__ w1, const float* __restrict__ w2,
                  _Float16* __restrict__ c1r, _Float16* __restrict__ c1i,
                  _Float16* __restrict__ c2r, _Float16* __restrict__ c2i)
{
  int idx = blockIdx.x * 256 + threadIdx.x;
  if (idx >= cNB * cCATK * cBS) return;
  int o = idx % cBS, i = (idx / cBS) % cCATK, n = idx / (cBS * cCATK);
  int i2 = (i < cBS) ? i : (i - cBS);
  float a0 = w1[((0 * cNB + n) * cBS + i2) * cBS + o];
  float a1 = w1[((1 * cNB + n) * cBS + i2) * cBS + o];
  c1r[idx] = (_Float16)((i < cBS) ? a0 : -a1);   // o_r = [xr xi] @ [w10; -w11]
  c1i[idx] = (_Float16)((i < cBS) ? a1 :  a0);   // o_i = [xr xi] @ [w11;  w10]
  float b0 = w2[((0 * cNB + n) * cBS + i2) * cBS + o];
  float b1 = w2[((1 * cNB + n) * cBS + i2) * cBS + o];
  c2r[idx] = (_Float16)((i < cBS) ? b0 : -b1);
  c2i[idx] = (_Float16)((i < cBS) ? b1 :  b0);
}

__global__ __launch_bounds__(256)
void k_f32_to_f16(const float* __restrict__ s, _Float16* __restrict__ d, int n)
{
  int i = blockIdx.x * 256 + threadIdx.x;
  if (i < n) d[i] = (_Float16)s[i];
}

// Gather patch rows: A[token][c*256 + p*16 + q] = img[c][h*16+p][w*16+q] (f16)
__global__ __launch_bounds__(256)
void k_patch_gather(const float* __restrict__ cst, const float* __restrict__ prog,
                    _Float16* __restrict__ Ap)
{
  int idx = blockIdx.x * 256 + threadIdx.x;
  if (idx >= cTOK * cPK) return;
  int k = idx % cPK, tok = idx / cPK;
  int c = k >> 8, pq = k & 255, p = pq >> 4, q = pq & 15;
  int h = tok / cWG, w = tok % cWG;
  long long src = (long long)(h * 16 + p) * cWI + (w * 16 + q);
  float v = (c < 4) ? cst[(long long)c * cHWI + src] : prog[src];
  Ap[idx] = (_Float16)v;
}

// Scatter head output [token][p*16+q] -> image [h*16+p][w*16+q]
__global__ __launch_bounds__(256)
void k_head_scatter(const float* __restrict__ chead, float* __restrict__ out)
{
  int idx = blockIdx.x * 256 + threadIdx.x;
  if (idx >= cTOK * 256) return;
  int tok = idx >> 8, pq = idx & 255, p = pq >> 4, q = pq & 15;
  int h = tok / cWG, w = tok % cWG;
  out[(long long)(h * 16 + p) * cWI + (w * 16 + q)] = chead[idx];
}

} // namespace

extern "C" void kernel_launch(void* const* d_in, const int* in_sizes, int n_in,
                              void* d_out, int out_size, void* d_ws, size_t ws_size,
                              hipStream_t stream)
{
  (void)in_sizes; (void)n_in; (void)out_size; (void)ws_size;

  auto in = [&](int i) { return (const float*)d_in[i]; };
  const float* constants  = in(0);
  const float* prognostic = in(1);
  const float* patch_w    = in(2);
  const float* patch_b    = in(3);
  const float* pos_embed  = in(4);
  const float* head_w     = in(5);

  // ---------------- workspace carve-out (256B aligned) --------------------
  char* wsb = (char*)d_ws;
  size_t off = 0;
  auto alloc = [&](size_t bytes) -> void* {
    void* p = wsb + off;
    off = (off + bytes + 255) & ~(size_t)255;
    return p;
  };
  float*    X      = (float*)   alloc(sizeof(float)    * (size_t)cTOK * cE);
  float*    XLN    = (float*)   alloc(sizeof(float)    * (size_t)cTOK * cE);
  _Float16* XLN16  = (_Float16*)alloc(sizeof(_Float16) * (size_t)cTOK * cE);
  _Float16* HID16  = (_Float16*)alloc(sizeof(_Float16) * (size_t)cTOK * cMLP);
  _Float16* APATCH = (_Float16*)alloc(sizeof(_Float16) * (size_t)cTOK * cPK);
  _Float16* X16    = (_Float16*)alloc(sizeof(_Float16) * (size_t)cTOK * cE);
  float*    YR     = (float*)   alloc(sizeof(float)    * (size_t)cPTS * cE);
  float*    YI     = (float*)   alloc(sizeof(float)    * (size_t)cPTS * cE);
  float*    UR     = (float*)   alloc(sizeof(float)    * (size_t)cPTS * cE);
  float*    UI     = (float*)   alloc(sizeof(float)    * (size_t)cPTS * cE);
  _Float16* XCAT   = (_Float16*)alloc(sizeof(_Float16) * (size_t)cPTSP * cNB * cCATK);
  _Float16* O1CAT  = (_Float16*)alloc(sizeof(_Float16) * (size_t)cPTSP * cNB * cCATK);
  float*    O2R    = (float*)   alloc(sizeof(float)    * (size_t)cPTSP * cE);
  float*    O2I    = (float*)   alloc(sizeof(float)    * (size_t)cPTSP * cE);
  float*    CHEAD  = (float*)   alloc(sizeof(float)    * (size_t)cTOK * 256);
  _Float16* WP16   = (_Float16*)alloc(sizeof(_Float16) * (size_t)cPK * cE);
  _Float16* WH16   = (_Float16*)alloc(sizeof(_Float16) * (size_t)cE * 256);
  _Float16* WF1    = (_Float16*)alloc(sizeof(_Float16) * (size_t)cE * cMLP);
  _Float16* WF2    = (_Float16*)alloc(sizeof(_Float16) * (size_t)cMLP * cE);
  _Float16* C1R    = (_Float16*)alloc(sizeof(_Float16) * (size_t)cNB * cCATK * cBS);
  _Float16* C1I    = (_Float16*)alloc(sizeof(_Float16) * (size_t)cNB * cCATK * cBS);
  _Float16* C2R    = (_Float16*)alloc(sizeof(_Float16) * (size_t)cNB * cCATK * cBS);
  _Float16* C2I    = (_Float16*)alloc(sizeof(_Float16) * (size_t)cNB * cCATK * cBS);
  float*    FWC    = (float*)   alloc(sizeof(float) * cKW * cWG);
  float*    FWS    = (float*)   alloc(sizeof(float) * cKW * cWG);
  float*    FHC    = (float*)   alloc(sizeof(float) * cHG * cHG);
  float*    FHS    = (float*)   alloc(sizeof(float) * cHG * cHG);
  float*    IWC    = (float*)   alloc(sizeof(float) * cKW * cWG);
  float*    IWS    = (float*)   alloc(sizeof(float) * cKW * cWG);

  auto blocks = [](long long n) { return dim3((unsigned)((n + 255) / 256)); };
  auto gemm = [&](const _Float16* A, const _Float16* B, float* Cf, _Float16* Ch,
                  const float* bias, const float* add,
                  int M, int N, int K, int lda, int ldb, int ldc, int ldadd,
                  long long sA, long long sB, long long sC, long long sBias,
                  int batch, int act) {
    dim3 g((N + 63) / 64, (M + 127) / 128, batch);
    k_wmma_gemm<<<g, 256, 0, stream>>>(A, B, Cf, Ch, bias, add, M, N, K,
                                       lda, ldb, ldc, ldadd, sA, sB, sC, sBias, act);
  };
  auto cvt = [&](const float* s, _Float16* d, long long n) {
    k_f32_to_f16<<<blocks(n), 256, 0, stream>>>(s, d, (int)n);
  };

  // ---------------- constant prep -----------------------------------------
  k_init_tables<<<blocks(cKW * cWG), 256, 0, stream>>>(FWC, FWS, FHC, FHS, IWC, IWS);
  cvt(patch_w, WP16, (long long)cPK * cE);
  cvt(head_w,  WH16, (long long)cE * 256);

  // ---------------- two autoregressive steps -------------------------------
  for (int s = 0; s < 2; ++s) {
    const float* prog = (s == 0) ? prognostic : (const float*)d_out;  // prev output
    k_patch_gather<<<blocks((long long)cTOK * cPK), 256, 0, stream>>>(constants, prog, APATCH);
    // patch embed + patch_b + pos_embed -> X
    gemm(APATCH, WP16, X, nullptr, patch_b, pos_embed,
         cTOK, cE, cPK, cPK, cE, cE, cE, 0, 0, 0, 0, 1, 0);

    for (int blk = 0; blk < 4; ++blk) {
      const int base = 6 + blk * 12;
      const float *ln1g = in(base + 0), *ln1b = in(base + 1);
      const float *w1   = in(base + 2), *b1   = in(base + 3);
      const float *w2   = in(base + 4), *b2   = in(base + 5);
      const float *ln2g = in(base + 6), *ln2b = in(base + 7);
      const float *fc1w = in(base + 8), *fc1b = in(base + 9);
      const float *fc2w = in(base + 10), *fc2b = in(base + 11);

      cvt(fc1w, WF1, (long long)cE * cMLP);
      cvt(fc2w, WF2, (long long)cMLP * cE);
      k_build_cats<<<blocks((long long)cNB * cCATK * cBS), 256, 0, stream>>>(w1, w2, C1R, C1I, C2R, C2I);

      k_layernorm<<<cTOK, 256, 0, stream>>>(X, ln1g, ln1b, XLN, nullptr);

      // AFNO: rDFT2 as dense transforms
      k_fft_w_fwd<<<blocks((long long)cPTS * cE), 256, 0, stream>>>(XLN, FWC, FWS, YR, YI);
      k_fft_h_fwd<<<blocks((long long)cPTS * cE), 256, 0, stream>>>(YR, YI, FHC, FHS, XCAT);

      // frequency block-MLP: batched WMMA GEMMs, batch = 16 channel blocks
      const long long ldF = cNB * cCATK;               // 1536
      gemm(XCAT, C1R, nullptr, O1CAT,       b1,              nullptr,
           cPTSP, cBS, cCATK, (int)ldF, cBS, (int)ldF, 0,
           cCATK, (long long)cCATK * cBS, cCATK, cBS, cNB, 1);
      gemm(XCAT, C1I, nullptr, O1CAT + cBS, b1 + cNB * cBS,  nullptr,
           cPTSP, cBS, cCATK, (int)ldF, cBS, (int)ldF, 0,
           cCATK, (long long)cCATK * cBS, cCATK, cBS, cNB, 1);
      gemm(O1CAT, C2R, O2R, nullptr,        b2,              nullptr,
           cPTSP, cBS, cCATK, (int)ldF, cBS, cE, 0,
           cCATK, (long long)cCATK * cBS, cBS, cBS, cNB, 3);
      gemm(O1CAT, C2I, O2I, nullptr,        b2 + cNB * cBS,  nullptr,
           cPTSP, cBS, cCATK, (int)ldF, cBS, cE, 0,
           cCATK, (long long)cCATK * cBS, cBS, cBS, cNB, 3);

      k_fft_h_inv<<<blocks((long long)cPTS * cE), 256, 0, stream>>>(O2R, O2I, FHC, FHS, UR, UI);
      // irDFT along W + AFNO bias(XLN) + residual(X) -> X
      k_fft_w_inv<<<blocks((long long)cTOK * cE), 256, 0, stream>>>(UR, UI, IWC, IWS, XLN, X);

      // MLP: LN2 -> fc1(gelu) f16 -> fc2 + residual -> X
      k_layernorm<<<cTOK, 256, 0, stream>>>(X, ln2g, ln2b, XLN, XLN16);
      gemm(XLN16, WF1, nullptr, HID16, fc1b, nullptr,
           cTOK, cMLP, cE, cE, cMLP, cMLP, 0, 0, 0, 0, 0, 1, 2);
      gemm(HID16, WF2, X, nullptr, fc2b, X,
           cTOK, cE, cMLP, cMLP, cE, cE, cE, 0, 0, 0, 0, 1, 0);
    }

    // head: [4050x768] @ [768x256], scatter into this step's output image
    cvt(X, X16, (long long)cTOK * cE);
    gemm(X16, WH16, CHEAD, nullptr, nullptr, nullptr,
         cTOK, 256, cE, cE, 256, 256, 0, 0, 0, 0, 0, 1, 0);
    k_head_scatter<<<blocks((long long)cTOK * 256), 256, 0, stream>>>(
        CHEAD, (float*)d_out + (long long)s * cHWI);
  }
}
